// MultiHeadSelfAttention_17025250361812
// MI455X (gfx1250) — matrix operand order repaired and checked by reading, not compiled, not run on hardware
//
#include <hip/hip_runtime.h>

// Problem constants (from reference)
#define NB     2
#define T_SEQ  2048
#define NH     16
#define HD     64
#define C_DIM  1024          // NH * HD
#define N_QKV  3072          // 3 * C_DIM
#define M_ROWS 4096          // NB * T_SEQ

typedef __attribute__((ext_vector_type(16))) __bf16 v16bf;
typedef __attribute__((ext_vector_type(4)))  __bf16 v4bf;
typedef __attribute__((ext_vector_type(8)))  float  v8f;
typedef __attribute__((ext_vector_type(4)))  float  v4f;

__device__ __forceinline__ int lane_id() { return (int)(threadIdx.x & 31u); }

__device__ __forceinline__ v8f wmma_bf16(v16bf a, v16bf b, v8f c) {
  // D = A(16x32) * B(32x16) + C, fp32 accumulate
  return __builtin_amdgcn_wmma_f32_16x16x32_bf16(false, a, false, b, (short)0, c,
                                                 false, false);
}

// ---- Fragment loaders (contiguous k order, same in A and B) -----------------------
// A 16x32 bf16: row = lane&15, kbase = (lane>>4)*16,
//               elem e -> K=kbase+e   (contiguous 16)
__device__ __forceinline__ v16bf load_a_bf16(const __bf16* __restrict__ A, int lda,
                                             int m0, int k0) {
  int l = lane_id();
  int row = l & 15;
  int kb  = (l >> 4) << 4;
  const __bf16* p = A + (size_t)(m0 + row) * lda + (k0 + kb);
  v16bf f;
#pragma unroll
  for (int e = 0; e < 8; ++e) f[e] = p[e];
#pragma unroll
  for (int e = 0; e < 8; ++e) f[8 + e] = p[8 + e];
  return f;
}

// B 32x16: col = lane&15, kbase = (lane>>4)*16, elem e -> K = kbase+e.
// Source S is bf16 row-major [N x lds_] holding B^T (B[k][n] = S[n][k])
// -> per-lane contiguous 16 bf16 (two b128 loads).
__device__ __forceinline__ v16bf load_bT_bf16(const __bf16* __restrict__ S, int lds_,
                                              int k0, int n0) {
  int l = lane_id();
  int col = l & 15;
  int kb  = (l >> 4) << 4;
  const __bf16* p = S + (size_t)(n0 + col) * lds_ + (k0 + kb);
  v16bf f;
#pragma unroll
  for (int e = 0; e < 16; ++e) f[e] = p[e];
  return f;
}

// ---- Kernel 0: x (fp32) -> bf16, vectorized --------------------------------
__global__ void __launch_bounds__(256)
cvt_x_kernel(const float* __restrict__ x, unsigned short* __restrict__ xg) {
  __bf16* xb = (__bf16*)xg;
  size_t i = ((size_t)blockIdx.x * 256 + threadIdx.x) * 4;
  v4f a = *(const v4f*)(x + i);
  v4bf b;
#pragma unroll
  for (int e = 0; e < 4; ++e) b[e] = (__bf16)a[e];
  *(v4bf*)(xb + i) = b;
}

// ---- Kernel 1: W [K x N] fp32  ->  W^T [N x K] bf16 ------------------------
// 32x32 tile staged through LDS with CDNA5 async global->LDS copies.
__global__ void __launch_bounds__(256)
transpose_cvt_kernel(const float* __restrict__ in, unsigned short* __restrict__ outg,
                     int K, int N) {
  __bf16* out = (__bf16*)outg;
  __shared__ float tile[32][33];                 // padded: conflict-free col reads
  int k0 = blockIdx.x * 32;
  int n0 = blockIdx.y * 32;
  int t  = (int)threadIdx.x;

  // async copy: each thread moves 16B (4 floats); 256 threads cover 32x32 tile
  int r  = t >> 3;                               // 0..31 (row within tile)
  int c4 = (t & 7) << 2;                         // 0,4,...,28
  unsigned ldsa = (unsigned)(uintptr_t)&tile[r][c4];
  unsigned long long ga =
      (unsigned long long)(uintptr_t)(in + (size_t)(k0 + r) * N + (n0 + c4));
  asm volatile("global_load_async_to_lds_b128 %0, %1, off"
               :: "v"(ldsa), "v"(ga) : "memory");
  asm volatile("s_wait_asynccnt 0x0" ::: "memory");
  __syncthreads();

  // transposed, converted, coalesced write-out
  int tx = t & 31;                               // k within tile
  int ty = t >> 5;                               // 0..7
#pragma unroll
  for (int i = 0; i < 4; ++i) {
    int n = ty + i * 8;
    out[(size_t)(n0 + n) * K + (k0 + tx)] = (__bf16)tile[tx][n];
  }
}

// ---- Kernel 2: QKV GEMM (bf16 WMMA) + bias + scatter -----------------------
// Q,K -> [B][H][T][64] bf16 ; V -> transposed [B][H][64][T] bf16
__global__ void __launch_bounds__(256)
qkv_gemm_kernel(const unsigned short* __restrict__ xg,
                const unsigned short* __restrict__ wtg,
                const float* __restrict__ bias,
                unsigned short* __restrict__ qg, unsigned short* __restrict__ kg,
                unsigned short* __restrict__ vtg) {
  const __bf16* xb = (const __bf16*)xg;
  const __bf16* wt = (const __bf16*)wtg;         // [N_QKV][C_DIM]
  __bf16* qb = (__bf16*)qg;
  __bf16* kb = (__bf16*)kg;
  __bf16* vt = (__bf16*)vtg;

  int wid = (int)(threadIdx.x >> 5);
  int m0 = blockIdx.x * 128 + (wid >> 1) * 32;   // 4 waves in M
  int n0 = blockIdx.y * 128 + (wid & 1) * 64;    // 2 waves in N

  v8f z = {0.f, 0.f, 0.f, 0.f, 0.f, 0.f, 0.f, 0.f};
  v8f acc[2][4] = {{z, z, z, z}, {z, z, z, z}};

  for (int k0 = 0; k0 < C_DIM; k0 += 32) {
    if (k0 + 64 < C_DIM)
      __builtin_prefetch(xb + (size_t)m0 * C_DIM + (k0 + 64), 0, 1);
    v16bf a0 = load_a_bf16(xb, C_DIM, m0, k0);
    v16bf a1 = load_a_bf16(xb, C_DIM, m0 + 16, k0);
#pragma unroll
    for (int j = 0; j < 4; ++j) {
      v16bf bj = load_bT_bf16(wt, C_DIM, k0, n0 + j * 16);
      acc[0][j] = wmma_bf16(a0, bj, acc[0][j]);
      acc[1][j] = wmma_bf16(a1, bj, acc[1][j]);
    }
  }

  int l = lane_id();
  int col  = l & 15;
  int rowb = (l >> 4) << 3;
#pragma unroll
  for (int j = 0; j < 4; ++j) {
    int n   = n0 + j * 16 + col;
    int mat = n >> 10;                           // 0:q 1:k 2:v
    int h   = (n & 1023) >> 6;
    int d   = n & 63;
    float bv = bias[n];
#pragma unroll
    for (int i = 0; i < 2; ++i) {
#pragma unroll
      for (int r = 0; r < 8; ++r) {
        int m  = m0 + i * 16 + rowb + r;         // m = b*T + t
        int bi = m >> 11;
        int tt = m & (T_SEQ - 1);
        __bf16 val = (__bf16)(acc[i][j][r] + bv);
        if (mat == 0) {
          qb[(((size_t)(bi * NH + h)) * T_SEQ + tt) * HD + d] = val;
        } else if (mat == 1) {
          kb[(((size_t)(bi * NH + h)) * T_SEQ + tt) * HD + d] = val;
        } else {
          vt[(((size_t)(bi * NH + h)) * HD + d) * T_SEQ + tt] = val;   // V^T
        }
      }
    }
  }
}

// ---- Kernel 3: causal flash attention, 32 queries per wave -----------------
__global__ void __launch_bounds__(128)
attn_kernel(const unsigned short* __restrict__ qg,
            const unsigned short* __restrict__ kg,
            const unsigned short* __restrict__ vtg,
            unsigned short* __restrict__ og) {
  const __bf16* qbp = (const __bf16*)qg;
  const __bf16* kbp = (const __bf16*)kg;
  const __bf16* vtp = (const __bf16*)vtg;
  __bf16* ob = (__bf16*)og;

  __shared__ __bf16 plds[4][2 * 16 * 32];        // two 16x32 P tiles per wave

  int wid = (int)(threadIdx.x >> 5);
  int g   = blockIdx.x * 4 + wid;                // 0 .. 2047
  int t32 = g & 63;                              // T/32 = 64 query groups
  int h   = (g >> 6) & 15;
  int b   = g >> 10;
  int qbase = t32 * 32;

  size_t ho = ((size_t)(b * NH + h)) * T_SEQ * HD;
  const __bf16* Q  = qbp + ho;                   // [T][64]
  const __bf16* K  = kbp + ho;                   // [T][64]
  const __bf16* VT = vtp + ho;                   // [64][T]

  v16bf qf[2][2];
  qf[0][0] = load_a_bf16(Q, HD, qbase, 0);
  qf[0][1] = load_a_bf16(Q, HD, qbase, 32);
  qf[1][0] = load_a_bf16(Q, HD, qbase + 16, 0);
  qf[1][1] = load_a_bf16(Q, HD, qbase + 16, 32);

  v8f z = {0.f, 0.f, 0.f, 0.f, 0.f, 0.f, 0.f, 0.f};
  v8f o[2][4] = {{z, z, z, z}, {z, z, z, z}};
  float mrow[2][8], lrow[2][8];
#pragma unroll
  for (int qi = 0; qi < 2; ++qi)
#pragma unroll
    for (int r = 0; r < 8; ++r) { mrow[qi][r] = -__builtin_inff(); lrow[qi][r] = 0.f; }

  int l = lane_id();
  int col  = l & 15;
  int rowb = (l >> 4) << 3;
  __bf16* pl = plds[wid];
  const float scale = 0.125f;                    // 1/sqrt(64)

  for (int kc = 0; kc < qbase + 32; kc += 32) {
    // K^T fragments shared by both query tiles
    v16bf kb0 = load_bT_bf16(K, HD, 0,  kc);
    v16bf kb1 = load_bT_bf16(K, HD, 32, kc);
    v16bf kb2 = load_bT_bf16(K, HD, 0,  kc + 16);
    v16bf kb3 = load_bT_bf16(K, HD, 32, kc + 16);

    float alpha[2][8];
#pragma unroll
    for (int qi = 0; qi < 2; ++qi) {
      v8f s0 = z, s1 = z;
      s0 = wmma_bf16(qf[qi][0], kb0, s0);
      s0 = wmma_bf16(qf[qi][1], kb1, s0);
      s1 = wmma_bf16(qf[qi][0], kb2, s1);
      s1 = wmma_bf16(qf[qi][1], kb3, s1);

      float rm[8];
#pragma unroll
      for (int r = 0; r < 8; ++r) {
        int q = qbase + qi * 16 + rowb + r;
        float v0 = s0[r] * scale;
        float v1 = s1[r] * scale;
        if (kc + col > q)      v0 = -__builtin_inff();
        if (kc + 16 + col > q) v1 = -__builtin_inff();
        s0[r] = v0; s1[r] = v1;
        rm[r] = fmaxf(v0, v1);
      }
#pragma unroll
      for (int off = 1; off < 16; off <<= 1) {
#pragma unroll
        for (int r = 0; r < 8; ++r) rm[r] = fmaxf(rm[r], __shfl_xor(rm[r], off, 32));
      }

      float p0[8], p1[8], rs[8];
#pragma unroll
      for (int r = 0; r < 8; ++r) {
        float mn = fmaxf(mrow[qi][r], rm[r]);
        alpha[qi][r] = __expf(mrow[qi][r] - mn);
        mrow[qi][r]  = mn;
        p0[r] = __expf(s0[r] - mn);
        p1[r] = __expf(s1[r] - mn);
        rs[r] = p0[r] + p1[r];
      }
#pragma unroll
      for (int off = 1; off < 16; off <<= 1) {
#pragma unroll
        for (int r = 0; r < 8; ++r) rs[r] += __shfl_xor(rs[r], off, 32);
      }
#pragma unroll
      for (int r = 0; r < 8; ++r) lrow[qi][r] = lrow[qi][r] * alpha[qi][r] + rs[r];

      // rescale O for this query tile, stage P into LDS
#pragma unroll
      for (int n = 0; n < 4; ++n)
#pragma unroll
        for (int r = 0; r < 8; ++r) o[qi][n][r] *= alpha[qi][r];
      __bf16* pq = pl + qi * 512;
#pragma unroll
      for (int r = 0; r < 8; ++r) {
        pq[(rowb + r) * 32 + col]      = (__bf16)p0[r];
        pq[(rowb + r) * 32 + 16 + col] = (__bf16)p1[r];
      }
    }

    asm volatile("s_wait_dscnt 0x0" ::: "memory");
    v16bf pf0 = load_a_bf16(pl, 32, 0, 0);
    v16bf pf1 = load_a_bf16(pl + 512, 32, 0, 0);

    // O += P · V  (V^T gives contiguous B fragments)
#pragma unroll
    for (int n = 0; n < 4; ++n) {
      v16bf vf = load_bT_bf16(VT, T_SEQ, kc, n * 16);
      o[0][n] = wmma_bf16(pf0, vf, o[0][n]);
      o[1][n] = wmma_bf16(pf1, vf, o[1][n]);
    }
  }

  // normalize + store attention output as bf16 [B*T][C]
  size_t mb = (size_t)(b * T_SEQ + qbase);
#pragma unroll
  for (int qi = 0; qi < 2; ++qi) {
#pragma unroll
    for (int n = 0; n < 4; ++n) {
#pragma unroll
      for (int r = 0; r < 8; ++r) {
        size_t m = mb + qi * 16 + rowb + r;
        ob[m * C_DIM + h * HD + n * 16 + col] = (__bf16)(o[qi][n][r] / lrow[qi][r]);
      }
    }
  }
}

// ---- Kernel 4: output projection GEMM + bias -------------------------------
__global__ void __launch_bounds__(256)
proj_gemm_kernel(const unsigned short* __restrict__ ag,
                 const unsigned short* __restrict__ wtg,
                 const float* __restrict__ bias, float* __restrict__ out) {
  const __bf16* A  = (const __bf16*)ag;          // [M_ROWS][C_DIM]
  const __bf16* wt = (const __bf16*)wtg;         // [C_DIM][C_DIM] (W^T)

  int wid = (int)(threadIdx.x >> 5);
  int m0 = blockIdx.x * 128 + (wid >> 1) * 32;
  int n0 = blockIdx.y * 128 + (wid & 1) * 64;

  v8f z = {0.f, 0.f, 0.f, 0.f, 0.f, 0.f, 0.f, 0.f};
  v8f acc[2][4] = {{z, z, z, z}, {z, z, z, z}};

  for (int k0 = 0; k0 < C_DIM; k0 += 32) {
    if (k0 + 64 < C_DIM)
      __builtin_prefetch(A + (size_t)m0 * C_DIM + (k0 + 64), 0, 1);
    v16bf a0 = load_a_bf16(A, C_DIM, m0, k0);
    v16bf a1 = load_a_bf16(A, C_DIM, m0 + 16, k0);
#pragma unroll
    for (int j = 0; j < 4; ++j) {
      v16bf bj = load_bT_bf16(wt, C_DIM, k0, n0 + j * 16);
      acc[0][j] = wmma_bf16(a0, bj, acc[0][j]);
      acc[1][j] = wmma_bf16(a1, bj, acc[1][j]);
    }
  }

  int l = lane_id();
  int col  = l & 15;
  int rowb = (l >> 4) << 3;
#pragma unroll
  for (int j = 0; j < 4; ++j) {
    int n = n0 + j * 16 + col;
    float bv = bias[n];
#pragma unroll
    for (int i = 0; i < 2; ++i) {
#pragma unroll
      for (int r = 0; r < 8; ++r) {
        int m = m0 + i * 16 + rowb + r;
        out[(size_t)m * C_DIM + n] = acc[i][j][r] + bv;
      }
    }
  }
}

// ---- Host launcher ---------------------------------------------------------
extern "C" void kernel_launch(void* const* d_in, const int* in_sizes, int n_in,
                              void* d_out, int out_size, void* d_ws, size_t ws_size,
                              hipStream_t stream) {
  const float* x      = (const float*)d_in[0];
  const float* w_qkv  = (const float*)d_in[1];
  const float* b_qkv  = (const float*)d_in[2];
  const float* w_proj = (const float*)d_in[3];
  const float* b_proj = (const float*)d_in[4];
  float* out = (float*)d_out;

  char* ws = (char*)d_ws;
  const size_t SZ_X   = (size_t)M_ROWS * C_DIM * 2;   // 8 MB (bf16)
  const size_t SZ_WQ  = (size_t)N_QKV * C_DIM * 2;    // 6 MB
  const size_t SZ_WP  = (size_t)C_DIM * C_DIM * 2;    // 2 MB
  unsigned short* xb   = (unsigned short*)(ws);                         // 8 MB
  unsigned short* wqkt = (unsigned short*)(ws + SZ_X);                  // 6 MB
  unsigned short* wpjt = (unsigned short*)(ws + SZ_X + SZ_WQ);          // 2 MB
  unsigned short* qg   = (unsigned short*)(ws + SZ_X + SZ_WQ + SZ_WP);  // 8 MB
  unsigned short* kg   = (unsigned short*)((char*)qg + SZ_X);           // 8 MB
  unsigned short* vtg  = (unsigned short*)((char*)kg + SZ_X);           // 8 MB
  unsigned short* ag   = xb;   // reuse: xb is dead after the QKV GEMM

  cvt_x_kernel<<<dim3((M_ROWS * C_DIM) / (256 * 4)), 256, 0, stream>>>(x, xb);
  transpose_cvt_kernel<<<dim3(C_DIM / 32, N_QKV / 32), 256, 0, stream>>>(
      w_qkv, wqkt, C_DIM, N_QKV);
  transpose_cvt_kernel<<<dim3(C_DIM / 32, C_DIM / 32), 256, 0, stream>>>(
      w_proj, wpjt, C_DIM, C_DIM);
  qkv_gemm_kernel<<<dim3(M_ROWS / 128, N_QKV / 128), 256, 0, stream>>>(
      xb, wqkt, b_qkv, qg, kg, vtg);
  attn_kernel<<<dim3((NB * NH * (T_SEQ / 32)) / 4), 128, 0, stream>>>(
      qg, kg, vtg, ag);
  proj_gemm_kernel<<<dim3(M_ROWS / 128, C_DIM / 128), 256, 0, stream>>>(
      ag, wpjt, b_proj, out);
}
